// Model_18064632447276
// MI455X (gfx1250) — compile-verified, matrix operand order checked by reading, not run on hardware
//
#include <hip/hip_runtime.h>
#include <hip/hip_bf16.h>

// ---------------------------------------------------------------------------
// Model dims: V=128000, D=300, H=256, P=11, R=2048, TO=4096, TR=131072.
// 3H=768, 2H=512.  Padded: D 300->320, P 11->32, S(gp) 2049->2064, S(gc) 1->16.
// ---------------------------------------------------------------------------

typedef __attribute__((ext_vector_type(16))) __bf16 v16bf;
typedef __attribute__((ext_vector_type(8)))  __bf16 v8bf;
typedef __attribute__((ext_vector_type(8)))  float  v8f;

#define GRIDSTRIDE(i, total) \
  for (size_t i = (size_t)blockIdx.x * blockDim.x + threadIdx.x; i < (total); \
       i += (size_t)gridDim.x * blockDim.x)

// ------------------------------ utility kernels ----------------------------

__global__ void zero_f32_k(float* p, size_t n) {
  GRIDSTRIDE(i, n) p[i] = 0.0f;
}

// Pad-convert f32 [rows, sc] -> bf16 [rows, dc] (cols >= sc zero-filled).
__global__ void cvt_pad_bf16_k(const float* __restrict__ src,
                               __bf16* __restrict__ dst,
                               int rows, int sc, int dc) {
  size_t total = (size_t)rows * dc;
  GRIDSTRIDE(i, total) {
    int c = (int)(i % dc);
    size_t r = i / dc;
    dst[i] = (c < sc) ? (__bf16)src[r * sc + c] : (__bf16)0.0f;
  }
}

// Wh [768,256] -> WhT [256,768] so recurrence loads are lane-coalesced.
__global__ void transpose_wh_k(const float* __restrict__ src,
                               float* __restrict__ dst) {
  GRIDSTRIDE(i, (size_t)768 * 256) {
    int r = (int)(i >> 8);        // 0..767
    int c = (int)(i & 255);       // 0..255
    dst[(size_t)c * 768 + r] = src[i];
  }
}

// ------------------------------ embedding pooling --------------------------

__global__ void pool_orig_k(const float* __restrict__ embed,
                            const int* __restrict__ toks, int n_tok,
                            float* __restrict__ dst) {
  int d = blockIdx.x * blockDim.x + threadIdx.x;
  if (d >= 300) return;
  float s = 0.0f;
  for (int t = 0; t < n_tok; ++t)
    s += embed[(size_t)toks[t] * 300 + d];
  dst[d] = s;
}

__global__ void pool_reply_k(const float* __restrict__ embed,
                             const int* __restrict__ toks,
                             const int* __restrict__ seg, int n_tok,
                             float* __restrict__ dst) {
  size_t total = (size_t)n_tok * 300;
  GRIDSTRIDE(i, total) {
    size_t t = i / 300;
    int d = (int)(i % 300);
    atomicAdd(&dst[(size_t)seg[t] * 320 + d],
              embed[(size_t)toks[t] * 300 + d]);
  }
}

__global__ void build_xp_k(const float* __restrict__ op,
                           const float* __restrict__ rp,
                           float* __restrict__ xp) {
  GRIDSTRIDE(i, (size_t)2049 * 11) {
    int r = (int)(i / 11), c = (int)(i % 11);
    xp[(size_t)r * 32 + c] = (r == 0) ? op[c] : rp[(size_t)(r - 1) * 11 + c];
  }
}

// ------------------------------ WMMA GEMM ----------------------------------
// C[M,N] f32 = A[M,Kp] bf16 @ W[N,Kp]^T bf16 + bias[N]
// One wave per 16x64 C tile (4 accumulators): the A fragment is loaded ONCE
// per K-step and reused by 4 WMMAs -> 4x less A traffic, 4x WMMA density.
// grid = (N/64, M/16).  Fragment layout per CDNA5 ISA 7.12.2:
//   lane<16: row=lane, K = {k0..k0+7, k0+16..k0+23}   (two contiguous runs)
//   lane>=16: row=lane-16, K = {k0+8..k0+15, k0+24..k0+31}
__global__ void __launch_bounds__(32)
wmma_gemm_bf16_k(const __bf16* __restrict__ A, const __bf16* __restrict__ W,
                 const float* __restrict__ bias, float* __restrict__ C,
                 int M, int N, int Kp) {
  const int lane = threadIdx.x;
  const int half = lane >> 4;
  const int l16  = lane & 15;
  const int nbase = blockIdx.x * 64;
  const int mbase = blockIdx.y * 16;

  v8f acc[4];
#pragma unroll
  for (int j = 0; j < 4; ++j) {
    const float b = bias[nbase + j * 16 + l16];
#pragma unroll
    for (int i = 0; i < 8; ++i) acc[j][i] = b;
  }

  const __bf16* arow = A + (size_t)(mbase + l16) * Kp + half * 8;
  const __bf16* brow0 = W + (size_t)(nbase + l16) * Kp + half * 8;

  for (int k0 = 0; k0 < Kp; k0 += 32) {
    v8bf alo = *reinterpret_cast<const v8bf*>(arow + k0);
    v8bf ahi = *reinterpret_cast<const v8bf*>(arow + k0 + 16);
    v16bf av;
#pragma unroll
    for (int i = 0; i < 8; ++i) { av[i] = alo[i]; av[i + 8] = ahi[i]; }

#pragma unroll
    for (int j = 0; j < 4; ++j) {
      const __bf16* brow = brow0 + (size_t)j * 16 * Kp;
      v8bf blo = *reinterpret_cast<const v8bf*>(brow + k0);
      v8bf bhi = *reinterpret_cast<const v8bf*>(brow + k0 + 16);
      v16bf bv;
#pragma unroll
      for (int i = 0; i < 8; ++i) { bv[i] = blo[i]; bv[i + 8] = bhi[i]; }
      acc[j] = __builtin_amdgcn_wmma_f32_16x16x32_bf16(
          false, av, false, bv, (short)0, acc[j], false, false);
    }
  }

  // C/D layout: lanes 0-15 -> M = vgpr (0..7); lanes 16-31 -> M = 8+vgpr.
#pragma unroll
  for (int j = 0; j < 4; ++j) {
#pragma unroll
    for (int r = 0; r < 8; ++r) {
      int row = mbase + half * 8 + r;
      C[(size_t)row * N + nbase + j * 16 + l16] = acc[j][r];
    }
  }
}

// ------------------------------ GRU recurrence -----------------------------
// ALL branches and directions in ONE launch: grid = (2 dirs, 3 branches).
// The two ~2050-step sequential chains (reply, profile) now run concurrently
// on different WGPs instead of serializing on the stream.
// h lives in LDS; WhT read from L2 (786 KB f32 > 320 KB LDS; model fits L2).
__global__ void __launch_bounds__(256)
gru_all_k(const float* __restrict__ gi0, const float* __restrict__ gi1,
          const float* __restrict__ gi2,
          const float* __restrict__ WhT0, const float* __restrict__ WhT1,
          const float* __restrict__ WhT2,
          const float* __restrict__ bh0, const float* __restrict__ bh1,
          const float* __restrict__ bh2,
          float* __restrict__ out0, float* __restrict__ out1,
          float* __restrict__ out2,
          float* __restrict__ hl0, float* __restrict__ hl1,
          float* __restrict__ hl2) {
  const int dir = blockIdx.x;
  const int br  = blockIdx.y;
  const int t   = threadIdx.x;

  const int    S      = (br == 0) ? 1 : (br == 1) ? 2048 : 2049;
  const size_t gistrd = (br == 0) ? (size_t)16 * 768
                      : (br == 1) ? (size_t)2048 * 768 : (size_t)2064 * 768;
  const float* gi   = (br == 0) ? gi0  : (br == 1) ? gi1  : gi2;
  const float* WhT  = (br == 0) ? WhT0 : (br == 1) ? WhT1 : WhT2;
  const float* bh   = (br == 0) ? bh0  : (br == 1) ? bh1  : bh2;
  float*       out  = (br == 0) ? out0 : (br == 1) ? out1 : out2;
  float*       hl   = (br == 0) ? hl0  : (br == 1) ? hl1  : hl2;

  const float* giD = gi + (size_t)dir * gistrd;
  const float* Wt  = WhT + (size_t)dir * 256 * 768;
  const float* bhD = bh + (size_t)dir * 768;

  __shared__ float h[256];
  h[t] = 0.0f;
  __syncthreads();

  const float bhg0 = bhD[t], bhg1 = bhD[256 + t], bhg2 = bhD[512 + t];

  for (int s = 0; s < S; ++s) {
    const int j = (dir == 0) ? s : (S - 1 - s);
    const float* girow = giD + (size_t)j * 768;
    if (s + 1 < S) {  // prefetch next step's input gates (global_prefetch_b8)
      const int jn = (dir == 0) ? (s + 1) : (S - 2 - s);
      __builtin_prefetch(giD + (size_t)jn * 768 + t, 0, 0);
    }
    float g0 = bhg0, g1 = bhg1, g2 = bhg2;
#pragma unroll 4
    for (int k = 0; k < 256; ++k) {
      const float hk = h[k];
      const float* w = Wt + (size_t)k * 768;
      g0 = fmaf(w[t],       hk, g0);
      g1 = fmaf(w[256 + t], hk, g1);
      g2 = fmaf(w[512 + t], hk, g2);
    }
    const float r = 1.0f / (1.0f + __expf(-(girow[t] + g0)));
    const float z = 1.0f / (1.0f + __expf(-(girow[256 + t] + g1)));
    const float n = tanhf(girow[512 + t] + r * g2);
    const float hold = h[t];
    const float hnew = (1.0f - z) * n + z * hold;
    __syncthreads();
    h[t] = hnew;
    out[(size_t)j * 512 + dir * 256 + t] = hnew;
    __syncthreads();
  }
  hl[dir * 256 + t] = h[t];
}

// ------------------------------ attention ----------------------------------
// All 3 branches in one launch (grid.x = branch).
__global__ void __launch_bounds__(256)
attend_all_k(const float* __restrict__ outl0, const float* __restrict__ outl1,
             const float* __restrict__ outl2,
             const float* __restrict__ hl0, const float* __restrict__ hl1,
             const float* __restrict__ hl2,
             float* __restrict__ ctx, float* __restrict__ a_out) {
  const int br = blockIdx.x;
  const int t = threadIdx.x;
  const int S = (br == 0) ? 1 : (br == 1) ? 2048 : 2049;
  const float* outl  = (br == 0) ? outl0 : (br == 1) ? outl1 : outl2;
  const float* hlast = (br == 0) ? hl0   : (br == 1) ? hl1   : hl2;
  float* ctxp = ctx + br * 256;
  float* ao   = (br == 0) ? a_out : nullptr;

  __shared__ float q[256];
  __shared__ float sc[2049];
  __shared__ float red[256];

  q[t] = hlast[t] + hlast[256 + t];
  __syncthreads();

  for (int s = t; s < S; s += 256) {
    const float* row = outl + (size_t)s * 256;
    float d = 0.0f;
    for (int j = 0; j < 256; ++j) d = fmaf(row[j], q[j], d);
    sc[s] = d * 0.0625f;  // 1/sqrt(256)
  }
  __syncthreads();

  float mx = -1e30f;
  for (int s = t; s < S; s += 256) mx = fmaxf(mx, sc[s]);
  red[t] = mx;
  __syncthreads();
  for (int o = 128; o > 0; o >>= 1) {
    if (t < o) red[t] = fmaxf(red[t], red[t + o]);
    __syncthreads();
  }
  mx = red[0];
  __syncthreads();

  float sm = 0.0f;
  for (int s = t; s < S; s += 256) {
    const float e = __expf(sc[s] - mx);
    sc[s] = e;
    sm += e;
  }
  red[t] = sm;
  __syncthreads();
  for (int o = 128; o > 0; o >>= 1) {
    if (t < o) red[t] += red[t + o];
    __syncthreads();
  }
  const float inv = 1.0f / red[0];

  float c = 0.0f;
  for (int s = 0; s < S; ++s)
    c = fmaf(sc[s] * inv, outl[(size_t)s * 256 + t], c);
  ctxp[t] = c;

  if (ao != nullptr && t == 0)
    for (int s = 0; s < S; ++s) ao[s] = sc[s] * inv;
}

// ------------------------------ final head ---------------------------------
__global__ void __launch_bounds__(256)
final_k(const float* __restrict__ ctx, const float* __restrict__ outW,
        const float* __restrict__ outb, const float* __restrict__ label,
        const float* __restrict__ outl_c, const float* __restrict__ ad,
        float* __restrict__ out) {
  __shared__ float red[256];
  const int t = threadIdx.x;
  float d = 0.0f;
  for (int i = t; i < 768; i += 256) d = fmaf(ctx[i], outW[i], d);
  red[t] = d;
  __syncthreads();
  for (int o = 128; o > 0; o >>= 1) {
    if (t < o) red[t] += red[t + o];
    __syncthreads();
  }
  out[t] = outl_c[t];  // out_ori [1,1,256]
  if (t == 0) {
    const float res = 1.0f / (1.0f + __expf(-(red[0] + outb[0])));
    const float l = label[0] - res;
    out[256] = l * l;   // loss
    out[257] = res;     // result
    out[258] = ad[0];   // ad_ori (S=1)
  }
}

// ------------------------------ launch -------------------------------------

extern "C" void kernel_launch(void* const* d_in, const int* in_sizes, int n_in,
                              void* d_out, int out_size, void* d_ws,
                              size_t ws_size, hipStream_t stream) {
  (void)in_sizes; (void)n_in; (void)out_size; (void)ws_size;

  const float* embed   = (const float*)d_in[0];
  const float* gc_Wi   = (const float*)d_in[1];
  const float* gc_Wh   = (const float*)d_in[2];
  const float* gc_bi   = (const float*)d_in[3];
  const float* gc_bh   = (const float*)d_in[4];
  const float* gr_Wi   = (const float*)d_in[5];
  const float* gr_Wh   = (const float*)d_in[6];
  const float* gr_bi   = (const float*)d_in[7];
  const float* gr_bh   = (const float*)d_in[8];
  const float* gp_Wi   = (const float*)d_in[9];
  const float* gp_Wh   = (const float*)d_in[10];
  const float* gp_bi   = (const float*)d_in[11];
  const float* gp_bh   = (const float*)d_in[12];
  const float* linW    = (const float*)d_in[13];
  const float* linb    = (const float*)d_in[14];
  const float* outW    = (const float*)d_in[15];
  const float* outb    = (const float*)d_in[16];
  const int*   o_toks  = (const int*)d_in[17];
  const int*   r_toks  = (const int*)d_in[18];
  const int*   r_seg   = (const int*)d_in[19];
  const float* o_prof  = (const float*)d_in[20];
  const float* r_prof  = (const float*)d_in[21];
  const float* label   = (const float*)d_in[22];

  // ---- workspace sub-allocation (256B aligned) ----
  char* wp = (char*)d_ws;
  auto alloc = [&](size_t bytes) -> void* {
    void* p = (void*)wp;
    wp += (bytes + 255) & ~(size_t)255;
    return p;
  };
  float*  reply_x  = (float*)alloc((size_t)2048 * 320 * 4);
  float*  Ac_f     = (float*)alloc((size_t)16 * 320 * 4);
  float*  xp_f     = (float*)alloc((size_t)2064 * 32 * 4);
  __bf16* Ar_bf    = (__bf16*)alloc((size_t)2048 * 320 * 2);
  __bf16* Ac_bf    = (__bf16*)alloc((size_t)16 * 320 * 2);
  __bf16* Ap_bf    = (__bf16*)alloc((size_t)2064 * 32 * 2);
  __bf16* Wic_bf   = (__bf16*)alloc((size_t)2 * 768 * 320 * 2);
  __bf16* Wir_bf   = (__bf16*)alloc((size_t)2 * 768 * 320 * 2);
  __bf16* Wip_bf   = (__bf16*)alloc((size_t)2 * 768 * 32 * 2);
  __bf16* linW_bf  = (__bf16*)alloc((size_t)256 * 512 * 2);
  float*  gi_c     = (float*)alloc((size_t)2 * 16 * 768 * 4);
  float*  gi_r     = (float*)alloc((size_t)2 * 2048 * 768 * 4);
  float*  gi_p     = (float*)alloc((size_t)2 * 2064 * 768 * 4);
  float*  WhT_c    = (float*)alloc((size_t)2 * 256 * 768 * 4);
  float*  WhT_r    = (float*)alloc((size_t)2 * 256 * 768 * 4);
  float*  WhT_p    = (float*)alloc((size_t)2 * 256 * 768 * 4);
  float*  out_c    = (float*)alloc((size_t)16 * 512 * 4);
  float*  out_r    = (float*)alloc((size_t)2048 * 512 * 4);
  float*  out_p    = (float*)alloc((size_t)2064 * 512 * 4);
  __bf16* outc_bf  = (__bf16*)alloc((size_t)16 * 512 * 2);
  __bf16* outr_bf  = (__bf16*)alloc((size_t)2048 * 512 * 2);
  __bf16* outp_bf  = (__bf16*)alloc((size_t)2064 * 512 * 2);
  float*  outl_c   = (float*)alloc((size_t)16 * 256 * 4);
  float*  outl_r   = (float*)alloc((size_t)2048 * 256 * 4);
  float*  outl_p   = (float*)alloc((size_t)2064 * 256 * 4);
  float*  hlast_c  = (float*)alloc((size_t)2 * 256 * 4);
  float*  hlast_r  = (float*)alloc((size_t)2 * 256 * 4);
  float*  hlast_p  = (float*)alloc((size_t)2 * 256 * 4);
  float*  ctx      = (float*)alloc((size_t)3 * 256 * 4);
  float*  ad       = (float*)alloc(4);

  // ---- zero accumulated / padded f32 buffers ----
  zero_f32_k<<<512, 256, 0, stream>>>(reply_x, (size_t)2048 * 320);
  zero_f32_k<<<32, 256, 0, stream>>>(Ac_f, (size_t)16 * 320);
  zero_f32_k<<<64, 256, 0, stream>>>(xp_f, (size_t)2064 * 32);
  zero_f32_k<<<32, 256, 0, stream>>>(out_c, (size_t)16 * 512);
  zero_f32_k<<<512, 256, 0, stream>>>(out_r, (size_t)2048 * 512);
  zero_f32_k<<<512, 256, 0, stream>>>(out_p, (size_t)2064 * 512);

  // ---- embedding pooling ----
  pool_orig_k<<<2, 256, 0, stream>>>(embed, o_toks, 4096, Ac_f);
  pool_reply_k<<<4096, 256, 0, stream>>>(embed, r_toks, r_seg, 131072, reply_x);
  build_xp_k<<<128, 256, 0, stream>>>(o_prof, r_prof, xp_f);

  // ---- bf16 conversions (pad K) ----
  cvt_pad_bf16_k<<<1024, 256, 0, stream>>>(gc_Wi, Wic_bf, 1536, 300, 320);
  cvt_pad_bf16_k<<<1024, 256, 0, stream>>>(gr_Wi, Wir_bf, 1536, 300, 320);
  cvt_pad_bf16_k<<<256, 256, 0, stream>>>(gp_Wi, Wip_bf, 1536, 11, 32);
  cvt_pad_bf16_k<<<512, 256, 0, stream>>>(linW, linW_bf, 256, 512, 512);
  cvt_pad_bf16_k<<<1024, 256, 0, stream>>>(reply_x, Ar_bf, 2048, 320, 320);
  cvt_pad_bf16_k<<<32, 256, 0, stream>>>(Ac_f, Ac_bf, 16, 320, 320);
  cvt_pad_bf16_k<<<128, 256, 0, stream>>>(xp_f, Ap_bf, 2064, 32, 32);

  // ---- input-gate GEMMs: gi = A @ Wi^T + bi (16x64 tile per wave) ----
  for (int d = 0; d < 2; ++d) {
    wmma_gemm_bf16_k<<<dim3(12, 1), 32, 0, stream>>>(
        Ac_bf, Wic_bf + (size_t)d * 768 * 320, gc_bi + (size_t)d * 768,
        gi_c + (size_t)d * 16 * 768, 16, 768, 320);
    wmma_gemm_bf16_k<<<dim3(12, 128), 32, 0, stream>>>(
        Ar_bf, Wir_bf + (size_t)d * 768 * 320, gr_bi + (size_t)d * 768,
        gi_r + (size_t)d * 2048 * 768, 2048, 768, 320);
    wmma_gemm_bf16_k<<<dim3(12, 129), 32, 0, stream>>>(
        Ap_bf, Wip_bf + (size_t)d * 768 * 32, gp_bi + (size_t)d * 768,
        gi_p + (size_t)d * 2064 * 768, 2064, 768, 32);
  }

  // ---- Wh transposes for coalesced recurrence loads ----
  for (int d = 0; d < 2; ++d) {
    transpose_wh_k<<<768, 256, 0, stream>>>(gc_Wh + (size_t)d * 768 * 256,
                                            WhT_c + (size_t)d * 256 * 768);
    transpose_wh_k<<<768, 256, 0, stream>>>(gr_Wh + (size_t)d * 768 * 256,
                                            WhT_r + (size_t)d * 256 * 768);
    transpose_wh_k<<<768, 256, 0, stream>>>(gp_Wh + (size_t)d * 768 * 256,
                                            WhT_p + (size_t)d * 256 * 768);
  }

  // ---- all bi-GRU recurrences in ONE launch: grid (2 dirs, 3 branches) ----
  gru_all_k<<<dim3(2, 3), 256, 0, stream>>>(
      gi_c, gi_r, gi_p, WhT_c, WhT_r, WhT_p, gc_bh, gr_bh, gp_bh,
      out_c, out_r, out_p, hlast_c, hlast_r, hlast_p);

  // ---- linear layer GEMMs: out_l = out @ linW^T + linb ----
  cvt_pad_bf16_k<<<32, 256, 0, stream>>>(out_c, outc_bf, 16, 512, 512);
  cvt_pad_bf16_k<<<1024, 256, 0, stream>>>(out_r, outr_bf, 2048, 512, 512);
  cvt_pad_bf16_k<<<1024, 256, 0, stream>>>(out_p, outp_bf, 2064, 512, 512);
  wmma_gemm_bf16_k<<<dim3(4, 1), 32, 0, stream>>>(outc_bf, linW_bf, linb,
                                                  outl_c, 16, 256, 512);
  wmma_gemm_bf16_k<<<dim3(4, 128), 32, 0, stream>>>(outr_bf, linW_bf, linb,
                                                    outl_r, 2048, 256, 512);
  wmma_gemm_bf16_k<<<dim3(4, 129), 32, 0, stream>>>(outp_bf, linW_bf, linb,
                                                    outl_p, 2064, 256, 512);

  // ---- attention, all branches in one launch ----
  attend_all_k<<<3, 256, 0, stream>>>(outl_c, outl_r, outl_p,
                                      hlast_c, hlast_r, hlast_p, ctx, ad);

  // ---- final head ----
  final_k<<<1, 256, 0, stream>>>(ctx, outW, outb, label, outl_c, ad,
                                 (float*)d_out);
}
